// SelfAttention_10230612099939
// MI455X (gfx1250) — compile-verified, matrix operand order checked by reading, not run on hardware
//
#include <hip/hip_runtime.h>
#include <cstdint>

#define B_   4
#define C_   256
#define N_   4096
#define G_   8
#define CPG_ 32
#define BC_  64          // key block per attention iteration

typedef __attribute__((ext_vector_type(16))) __bf16 v16bf;
typedef __attribute__((ext_vector_type(8)))  __bf16 v8bf;
typedef __attribute__((ext_vector_type(8)))  float  v8f;
typedef __attribute__((ext_vector_type(8)))  unsigned short u16x8;
typedef __attribute__((ext_vector_type(4)))  unsigned int v4u;
typedef __attribute__((ext_vector_type(8)))  int v8i;
typedef __attribute__((ext_vector_type(4)))  int v4i;

__device__ __forceinline__ unsigned short f2bf(float f) {
  unsigned u = __builtin_bit_cast(unsigned, f);
  u += 0x7fffu + ((u >> 16) & 1u);          // round-to-nearest-even
  return (unsigned short)(u >> 16);
}

__device__ __forceinline__ v8f zero8() {
  v8f z;
#pragma unroll
  for (int i = 0; i < 8; ++i) z[i] = 0.0f;
  return z;
}

// Fetch a 16x32 bf16 WMMA A/B operand tile from row-major [row][K] storage
// (leading dim `ld` elements). Per ISA 7.12.2: lane holds row (lane&15),
// K-chunks at (lane>>4)*8 and 16+(lane>>4)*8  -> two 16B loads.
__device__ __forceinline__ v16bf load_ab_tile(const unsigned short* base, int row0,
                                              int ld, int k0, int lane) {
  int r  = row0 + (lane & 15);
  int kh = (lane >> 4) << 3;
  const unsigned short* p = base + (size_t)r * ld + (size_t)(k0 + kh);
  v8bf lo = *(const v8bf*)p;
  v8bf hi = *(const v8bf*)(p + 16);
  return __builtin_shufflevector(lo, hi, 0,1,2,3,4,5,6,7,8,9,10,11,12,13,14,15);
}

__device__ __forceinline__ v8f wmma_bf16(v16bf a, v16bf b, v8f c) {
  return __builtin_amdgcn_wmma_f32_16x16x32_bf16(false, a, false, b, (short)0, c,
                                                 false, false);
}

// Issue one 2-D TDM load: tile_d1 rows x tile_d0 elements (bf16) from a
// row-major tensor (leading dim = stride0 elements) into dense LDS at lds_off.
// D# layout per CDNA5 ISA 8.3/8.4 (count=1, type=2, data_size=1 -> 2 bytes).
// This toolchain exposes the 6-arg builtin; the extra group operands are
// zero (2-D tile: tile_dim2/3/4 = 0).
__device__ __forceinline__ void tdm_load_2d(unsigned int lds_off, const void* gptr,
                                            unsigned tensor_d0, unsigned tensor_d1,
                                            unsigned tile_d0, unsigned tile_d1,
                                            unsigned stride0) {
  unsigned long long ga = (unsigned long long)(uintptr_t)gptr;
  v4u g0;
  g0[0] = 1u;                                             // count=1, user mode
  g0[1] = lds_off;                                        // lds_addr
  g0[2] = (unsigned)ga;                                   // global_addr[31:0]
  g0[3] = (unsigned)((ga >> 32) & 0x01FFFFFFu) | (2u << 30);  // addr[56:32], type=2
  v8i g1;
  g1[0] = (int)(1u << 16);                                // data_size=1 (2B)
  g1[1] = (int)(tensor_d0 << 16);                         // tensor_dim0[15:0]
  g1[2] = (int)((tensor_d0 >> 16) | (tensor_d1 << 16));   // d0 hi | d1 lo
  g1[3] = (int)((tensor_d1 >> 16) | (tile_d0 << 16));     // d1 hi | tile_dim0
  g1[4] = (int)(tile_d1 & 0xffffu);                       // tile_dim1, tile_dim2=0
  g1[5] = (int)stride0;                                   // tensor_dim0_stride lo32
  g1[6] = 0;
  g1[7] = 0;
  v4i z4;
  z4[0] = 0; z4[1] = 0; z4[2] = 0; z4[3] = 0;
  v8i z8;
#pragma unroll
  for (int i = 0; i < 8; ++i) z8[i] = 0;
  __builtin_amdgcn_tensor_load_to_lds(g0, g1, z4, z4, z8, 0);
}

// ---------------------------------------------------------------- weights->bf16
__global__ void k_prep_weights(const float* qw, const float* kw, const float* vw,
                               const float* pw, unsigned short* dst) {
  int i = blockIdx.x * blockDim.x + threadIdx.x;   // 0 .. 4*65536
  int m = i >> 16, e = i & 65535;
  const float* s = (m == 0) ? qw : (m == 1) ? kw : (m == 2) ? vw : pw;
  dst[i] = f2bf(s[e]);
}

// ---------------------------------------------------------------- GN stats
__global__ void k_gn_stats(const float* x, float* stat) {
  int bg = blockIdx.x;                             // b*8+g, 0..31
  const float* p = x + (size_t)bg * CPG_ * N_;
  float s = 0.f, s2 = 0.f;
  for (int i = threadIdx.x; i < CPG_ * N_; i += blockDim.x) {
    float v = p[i]; s += v; s2 += v * v;
  }
  __shared__ float rs[256], rq[256];
  rs[threadIdx.x] = s; rq[threadIdx.x] = s2;
  __syncthreads();
  for (int o = 128; o > 0; o >>= 1) {
    if ((int)threadIdx.x < o) {
      rs[threadIdx.x] += rs[threadIdx.x + o];
      rq[threadIdx.x] += rq[threadIdx.x + o];
    }
    __syncthreads();
  }
  if (threadIdx.x == 0) {
    float inv  = 1.0f / (float)(CPG_ * N_);
    float mean = rs[0] * inv;
    float var  = rq[0] * inv - mean * mean;
    stat[bg * 2 + 0] = mean;
    stat[bg * 2 + 1] = rsqrtf(var + 1e-5f);
  }
}

// ------------------------------------------------- GN apply + transpose to [n][c]
__global__ void k_gn_apply(const float* x, const float* gw, const float* gb,
                           const float* stat, unsigned short* h) {
  // grid (C_/32, N_/64, B_), 256 threads; blockIdx.x == group id (CPG_==32)
  int cb = blockIdx.x * 32, nb = blockIdx.y * 64, b = blockIdx.z;
  float mean = stat[(b * G_ + blockIdx.x) * 2 + 0];
  float rstd = stat[(b * G_ + blockIdx.x) * 2 + 1];
  __shared__ unsigned short tile[32][68];          // [c][n], padded
  const float* xb = x + ((size_t)b * C_ + cb) * N_ + nb;
#pragma unroll
  for (int i = 0; i < 8; ++i) {
    int idx = threadIdx.x + i * 256;
    int cl = idx >> 6, nl = idx & 63;              // coalesced along n
    float v = xb[(size_t)cl * N_ + nl];
    int c = cb + cl;
    v = (v - mean) * rstd * gw[c] + gb[c];
    tile[cl][nl] = f2bf(v);
  }
  __syncthreads();
  unsigned short* hb = h + ((size_t)b * N_ + nb) * C_ + cb;
#pragma unroll
  for (int i = 0; i < 8; ++i) {
    int idx = threadIdx.x + i * 256;
    int nl = idx >> 5, cl = idx & 31;              // coalesced along c
    hb[(size_t)nl * C_ + cl] = tile[cl][nl];
  }
}

// ---------------------------------------------------------------- QKV GEMMs
__global__ void __launch_bounds__(32) k_qkv(
    const unsigned short* h, const unsigned short* wq, const unsigned short* wk,
    const unsigned short* wv, const float* qb, const float* kb, const float* vb,
    unsigned short* q, unsigned short* k, unsigned short* v) {
  int lane = threadIdx.x;
  int nt = blockIdx.x, ot = blockIdx.y;
  int b = blockIdx.z / 3, which = blockIdx.z % 3;
  const unsigned short* w    = (which == 0) ? wq : (which == 1) ? wk : wv;
  const float*          bias = (which == 0) ? qb : (which == 1) ? kb : vb;
  const unsigned short* hb = h + (size_t)b * N_ * C_;

  v8f acc = zero8();
#pragma unroll
  for (int kk = 0; kk < C_; kk += 32) {
    v16bf a  = load_ab_tile(w,  ot * 16, C_, kk, lane);   // weights [o][c]
    v16bf bb = load_ab_tile(hb, nt * 16, C_, kk, lane);   // h [n][c]
    acc = wmma_bf16(a, bb, acc);
  }
  int hh = lane >> 4, lo = lane & 15;
  int obase = ot * 16 + hh * 8;
  const float4* bp = (const float4*)(bias + obase);
  float4 bq0 = bp[0], bq1 = bp[1];
  float bias8[8] = {bq0.x, bq0.y, bq0.z, bq0.w, bq1.x, bq1.y, bq1.z, bq1.w};
  int n = nt * 16 + lo;

  if (which == 2) {                                       // v -> [c][n]
    unsigned short* vd = v + (size_t)b * C_ * N_ + n;
#pragma unroll
    for (int r = 0; r < 8; ++r)
      vd[(size_t)(obase + r) * N_] = f2bf(acc[r] + bias8[r]);
  } else {                                                // q,k -> [n][c]
    float scale = (which == 0) ? 0.0625f : 1.0f;          // 1/sqrt(256) folded into q
    u16x8 outv;
#pragma unroll
    for (int r = 0; r < 8; ++r) outv[r] = f2bf((acc[r] + bias8[r]) * scale);
    unsigned short* dst = ((which == 0) ? q : k) + ((size_t)b * N_ + n) * C_ + obase;
    *(u16x8*)dst = outv;
  }
}

// ----------------------------------------------------------- flash attention
// Workgroup = 8 waves, each owns 16 query rows (128 rows/WG). Wave 0 streams
// K (64x256) and V (256x64) bf16 tiles into double-buffered LDS via TDM;
// all waves consume them. One barrier per key-block iteration.
__global__ void __launch_bounds__(256) k_attn(
    const unsigned short* q, const unsigned short* k, const unsigned short* v,
    unsigned short* o) {
  int lane = threadIdx.x & 31, wid = threadIdx.x >> 5;
  int b = blockIdx.y;
  int n0 = blockIdx.x * 128 + wid * 16;                   // 16 query rows per wave
  int hh = lane >> 4, lo = lane & 15;
  const unsigned short* qp = q + (size_t)b * N_ * C_;
  const unsigned short* kp = k + (size_t)b * N_ * C_;
  const unsigned short* vp = v + (size_t)b * C_ * N_;

  __shared__ unsigned short smk[2][BC_ * C_];             // K tiles [m][c]  2x32KB
  __shared__ unsigned short smv[2][C_ * BC_];             // V tiles [c][m]  2x32KB
  __shared__ unsigned short smp[8][16 * BC_];             // per-wave P buf  16KB
  unsigned short* pl = smp[wid];

  if (wid == 0) {                                         // prefetch buffer 0
    tdm_load_2d((unsigned)(uintptr_t)&smk[0][0], kp, C_, N_, C_, BC_, C_);
    tdm_load_2d((unsigned)(uintptr_t)&smv[0][0], vp, N_, C_, BC_, C_, N_);
  }

  v16bf qr[8];                                            // Q block in registers
#pragma unroll
  for (int kk = 0; kk < 8; ++kk) qr[kk] = load_ab_tile(qp, n0, C_, kk * 32, lane);

  v8f acc[16];                                            // O accum: 16n x 256c
#pragma unroll
  for (int t = 0; t < 16; ++t) acc[t] = zero8();
  float mrun[8], lrun[8];
#pragma unroll
  for (int r = 0; r < 8; ++r) { mrun[r] = -1e30f; lrun[r] = 0.0f; }

  for (int it = 0; it < N_ / BC_; ++it) {                 // 64 key blocks
    int buf = it & 1;
    __builtin_amdgcn_s_wait_tensorcnt(0);                 // TDM for `buf` done
    __syncthreads();                                      // publish; readers of
                                                          // buf^1 are also done
    if (wid == 0 && it + 1 < N_ / BC_) {                  // prefetch next block
      int m1 = (it + 1) * BC_;
      tdm_load_2d((unsigned)(uintptr_t)&smk[buf ^ 1][0], kp + (size_t)m1 * C_,
                  C_, N_, C_, BC_, C_);
      tdm_load_2d((unsigned)(uintptr_t)&smv[buf ^ 1][0], vp + m1,
                  N_, C_, BC_, C_, N_);
    }
    const unsigned short* kb = smk[buf];
    const unsigned short* vb = smv[buf];

    v8f s[4];                                             // S = Q^T K (16x64)
#pragma unroll
    for (int mt = 0; mt < 4; ++mt) s[mt] = zero8();
#pragma unroll
    for (int kk = 0; kk < 8; ++kk) {
#pragma unroll
      for (int mt = 0; mt < 4; ++mt) {
        v16bf bk = load_ab_tile(kb, mt * 16, C_, kk * 32, lane);
        s[mt] = wmma_bf16(qr[kk], bk, s[mt]);
      }
    }

    float mxr[8];                                         // per-row block max
#pragma unroll
    for (int r = 0; r < 8; ++r) {
      float mx = fmaxf(fmaxf(s[0][r], s[1][r]), fmaxf(s[2][r], s[3][r]));
#pragma unroll
      for (int off = 1; off < 16; off <<= 1) mx = fmaxf(mx, __shfl_xor(mx, off, 32));
      mxr[r] = mx;
    }
    int upd = 0;
#pragma unroll
    for (int r = 0; r < 8; ++r) upd |= (mxr[r] > mrun[r]) ? 1 : 0;
    if (__ballot(upd) != 0ull) {                          // lazy rescale (rare)
      float cf[8];
#pragma unroll
      for (int r = 0; r < 8; ++r) {
        float mnew = fmaxf(mrun[r], mxr[r]);
        cf[r] = __expf(mrun[r] - mnew);
        lrun[r] *= cf[r];
        mrun[r] = mnew;
      }
#pragma unroll
      for (int t = 0; t < 16; ++t)
#pragma unroll
        for (int r = 0; r < 8; ++r) acc[t][r] *= cf[r];
    }
#pragma unroll
    for (int r = 0; r < 8; ++r) {                         // P = exp(S - m)
      int nloc = r + hh * 8;
      float p0 = __expf(s[0][r] - mrun[r]);
      float p1 = __expf(s[1][r] - mrun[r]);
      float p2 = __expf(s[2][r] - mrun[r]);
      float p3 = __expf(s[3][r] - mrun[r]);
      pl[nloc * BC_ +      lo] = f2bf(p0);                // D-layout -> LDS [n][m]
      pl[nloc * BC_ + 16 + lo] = f2bf(p1);
      pl[nloc * BC_ + 32 + lo] = f2bf(p2);
      pl[nloc * BC_ + 48 + lo] = f2bf(p3);
      float rsum = (p0 + p1) + (p2 + p3);
#pragma unroll
      for (int off = 1; off < 16; off <<= 1) rsum += __shfl_xor(rsum, off, 32);
      lrun[r] += rsum;
    }
    v16bf pa0 = load_ab_tile(pl, 0, BC_, 0,  lane);       // P as A operand
    v16bf pa1 = load_ab_tile(pl, 0, BC_, 32, lane);
#pragma unroll
    for (int t = 0; t < 16; ++t) {                        // O += P * V
      v16bf bv0 = load_ab_tile(vb, t * 16, BC_, 0,  lane);
      v16bf bv1 = load_ab_tile(vb, t * 16, BC_, 32, lane);
      acc[t] = wmma_bf16(pa0, bv0, acc[t]);
      acc[t] = wmma_bf16(pa1, bv1, acc[t]);
    }
  }

  float inv[8];
#pragma unroll
  for (int r = 0; r < 8; ++r) inv[r] = 1.0f / lrun[r];
  unsigned short* ob = o + (size_t)b * N_ * C_;           // O -> [n][c] bf16
#pragma unroll
  for (int t = 0; t < 16; ++t) {
    int c = t * 16 + lo;
#pragma unroll
    for (int r = 0; r < 8; ++r) {
      int n = n0 + r + hh * 8;
      ob[(size_t)n * C_ + c] = f2bf(acc[t][r] * inv[r]);
    }
  }
}

// -------------------------------------------------- projection + residual add
__global__ void __launch_bounds__(32) k_proj(
    const unsigned short* o, const unsigned short* wp, const float* pb,
    const float* x, float* out) {
  int lane = threadIdx.x;
  int nt = blockIdx.x, ot = blockIdx.y, b = blockIdx.z;
  const unsigned short* ob = o + (size_t)b * N_ * C_;
  v8f acc = zero8();
#pragma unroll
  for (int kk = 0; kk < C_; kk += 32) {
    v16bf a  = load_ab_tile(wp, ot * 16, C_, kk, lane);   // pw [o][c]
    v16bf bb = load_ab_tile(ob, nt * 16, C_, kk, lane);   // O [n][c]
    acc = wmma_bf16(a, bb, acc);
  }
  int hh = lane >> 4, lo = lane & 15;
  int n = nt * 16 + lo;
  const float4* bp = (const float4*)(pb + ot * 16 + hh * 8);
  float4 b0 = bp[0], b1 = bp[1];
  float bias8[8] = {b0.x, b0.y, b0.z, b0.w, b1.x, b1.y, b1.z, b1.w};
#pragma unroll
  for (int r = 0; r < 8; ++r) {
    int oc = ot * 16 + hh * 8 + r;
    size_t idx = ((size_t)b * C_ + oc) * N_ + n;
    out[idx] = x[idx] + acc[r] + bias8[r];
  }
}

extern "C" void kernel_launch(void* const* d_in, const int* in_sizes, int n_in,
                              void* d_out, int out_size, void* d_ws, size_t ws_size,
                              hipStream_t stream) {
  (void)in_sizes; (void)n_in; (void)out_size; (void)ws_size;
  const float* x     = (const float*)d_in[0];
  const float* gnw   = (const float*)d_in[1];
  const float* gnb   = (const float*)d_in[2];
  const float* qw    = (const float*)d_in[3];
  const float* qbias = (const float*)d_in[4];
  const float* kw    = (const float*)d_in[5];
  const float* kbias = (const float*)d_in[6];
  const float* vw    = (const float*)d_in[7];
  const float* vbias = (const float*)d_in[8];
  const float* pw    = (const float*)d_in[9];
  const float* pbias = (const float*)d_in[10];
  float* out = (float*)d_out;

  char* w = (char*)d_ws;
  float*          stat = (float*)w;                         // 64 f32 (pad 1KB)
  unsigned short* wqb  = (unsigned short*)(w + 1024);       // 4 x 64K bf16
  unsigned short* wkb  = wqb + 65536;
  unsigned short* wvb  = wqb + 2 * 65536;
  unsigned short* wpb  = wqb + 3 * 65536;
  unsigned short* h    = (unsigned short*)(w + 1024 + 524288);
  const size_t    bnc  = (size_t)B_ * N_ * C_;
  unsigned short* qd = h  + bnc;
  unsigned short* kd = qd + bnc;
  unsigned short* vd = kd + bnc;
  unsigned short* od = vd + bnc;                            // total ~40.5 MB

  k_prep_weights<<<1024, 256, 0, stream>>>(qw, kw, vw, pw, wqb);
  k_gn_stats    <<<32, 256, 0, stream>>>(x, stat);
  k_gn_apply    <<<dim3(C_ / 32, N_ / 64, B_), 256, 0, stream>>>(x, gnw, gnb, stat, h);
  k_qkv         <<<dim3(N_ / 16, C_ / 16, 3 * B_), 32, 0, stream>>>(
                    h, wqb, wkb, wvb, qbias, kbias, vbias, qd, kd, vd);
  k_attn        <<<dim3(N_ / 128, B_), 256, 0, stream>>>(qd, kd, vd, od);
  k_proj        <<<dim3(N_ / 16, C_ / 16, B_), 32, 0, stream>>>(od, wpb, pbias, x, out);
}
